// CrossMLA_52836687675413
// MI455X (gfx1250) — matrix operand order repaired and checked by reading, not compiled, not run on hardware
//
#include <hip/hip_runtime.h>
#include <cstdint>

// ---------------- problem constants ----------------
#define B_     2
#define SQ_    2048
#define SKV_   2048
#define H_     16
#define D_     1024
#define DCQ_   256
#define QHD_   192      // Q_HEAD_DIM = 128 + 64
#define DPRE_  128
#define DROPE_ 64
#define DCKV_  16
#define KVDN_  80       // DC_KV + D_ROPE
#define MEGD_  256
#define DCV_   128
#define QUPN_  (H_*QHD_)   // 3072
#define KVUPN_ (H_*MEGD_)  // 4096
#define OUTK_  (H_*DCV_)   // 2048
#define BSQ_   (B_*SQ_)    // 4096
#define BSKV_  (B_*SKV_)   // 4096
#define EPS_   1e-7f
#define LN10K_ 9.210340371976184f
#define ASCALE_ 0.07216878364870323f  // 1/sqrt(192)

// Async global->LDS staging (gfx1250 GLOBAL_LOAD_ASYNC_TO_LDS_B128, ASYNCcnt).
// Set to 0 to fall back to plain VGPR-bounce staging if the assembler rejects it.
#ifndef USE_ASYNC_COPY
#define USE_ASYNC_COPY 1
#endif

typedef unsigned int   u32;
typedef unsigned short u16;
typedef __attribute__((ext_vector_type(16))) __bf16 bf16x16;
typedef __attribute__((ext_vector_type(8)))  float  f32x8;

union Frag { u32 u[8]; bf16x16 v; };

__device__ __forceinline__ u16 f2bf(float f) {
  u32 u = __builtin_bit_cast(u32, f);
  u32 r = (u + 0x7fffu + ((u >> 16) & 1u)) >> 16;
  return (u16)r;
}

__device__ __forceinline__ f32x8 wmma_bf16(const Frag& a, const Frag& b, f32x8 c) {
  return __builtin_amdgcn_wmma_f32_16x16x32_bf16(false, a.v, false, b.v,
                                                 (short)0, c, false, false);
}

// A fragment: 16(M)x32(K) bf16 from row-major [*, lda], tile origin pre-added.
// lane l: m=l&15, h=l>>4; v<4: K=h*8+2v ; v>=4: K=16+h*8+2(v-4)
__device__ __forceinline__ void load_a(Frag& f, const u16* A, int lda, int lane) {
  int m = lane & 15, h = lane >> 4;
  const u16* r = A + m * lda;
#pragma unroll
  for (int v = 0; v < 4; ++v) f.u[v]     = *(const u32*)(r + h * 8 + 2 * v);
#pragma unroll
  for (int v = 0; v < 4; ++v) f.u[4 + v] = *(const u32*)(r + 16 + h * 8 + 2 * v);
}

// B fragment: 32(K)x16(N) bf16 where storage is K-contiguous per N row
// (i.e. Bt is [N, ldb] row-major == B transposed). lane l: n=l&15, h=l>>4;
// VGPR v<4 holds K pair (h*8+2v, +1); v>=4 holds K pair (16+h*8+2(v-4), +1) for column n.
__device__ __forceinline__ void load_b(Frag& f, const u16* Bt, int ldb, int lane) {
  int n = lane & 15, h = lane >> 4;
  const u16* r = Bt + n * ldb;
#pragma unroll
  for (int v = 0; v < 8; ++v) f.u[v] = *(const u32*)(r + (v >> 2) * 16 + h * 8 + 2 * (v & 3));
}

// K=16 zero-padded variants (real K dim = 16, WMMA K = 32)
__device__ __forceinline__ void load_a_k16(Frag& f, const u16* A, int lda, int lane) {
  int m = lane & 15, h = lane >> 4;
  const u16* r = A + m * lda;
#pragma unroll
  for (int v = 0; v < 4; ++v) f.u[v] = *(const u32*)(r + h * 8 + 2 * v); // K<16 always
#pragma unroll
  for (int v = 0; v < 4; ++v) f.u[4 + v] = 0u;                           // K>=16
}
__device__ __forceinline__ void load_b_k16(Frag& f, const u16* Bt, int ldb, int lane) {
  int n = lane & 15, h = lane >> 4;
  const u16* r = Bt + n * ldb;
#pragma unroll
  for (int v = 0; v < 8; ++v) f.u[v] = (v < 4) ? *(const u32*)(r + h * 8 + 2 * v) : 0u;
}

// 16B async copy: global (64-bit addr) -> LDS (32-bit byte offset)
__device__ __forceinline__ void async_cp16(u32 lds_off, const u16* g) {
#if USE_ASYNC_COPY
  asm volatile("global_load_async_to_lds_b128 %0, %1, off"
               :: "v"(lds_off), "v"((unsigned long long)(uintptr_t)g)
               : "memory");
#endif
}

// ---------------- conversion kernels ----------------
__global__ void k_cvt_bf16(const float* __restrict__ x, u16* __restrict__ y, int n) {
  int i = blockIdx.x * 256 + threadIdx.x;
  if (i < n) y[i] = f2bf(x[i]);
}

// W is (K,N) f32 row-major -> Wt (N,K) bf16 row-major (K contiguous)
__global__ void k_cvt_tr(const float* __restrict__ W, u16* __restrict__ Wt, int K, int N) {
  int i = blockIdx.x * 256 + threadIdx.x;
  if (i >= K * N) return;
  int n = i / K, k = i - n * K;
  Wt[(size_t)n * K + k] = f2bf(W[(size_t)k * N + n]);
}

// ---------------- q down-proj + RMS norm ----------------
__global__ __launch_bounds__(256) void k_qdown(const u16* __restrict__ A,
                                               const u16* __restrict__ Wt,
                                               const float* __restrict__ nw,
                                               u16* __restrict__ qc) {
  __shared__ float ldsC[16][260];
  __shared__ float rowsq[16];
  int t = threadIdx.x, wave = t >> 5, lane = t & 31;
  int m0 = blockIdx.x * 16;
  int n0 = wave * 32;
  f32x8 c0 = {}, c1 = {};
  for (int k0 = 0; k0 < D_; k0 += 32) {
    Frag a, b0, b1;
    load_a(a, A + (size_t)m0 * D_ + k0, D_, lane);
    load_b(b0, Wt + (size_t)n0 * D_ + k0, D_, lane);
    load_b(b1, Wt + (size_t)(n0 + 16) * D_ + k0, D_, lane);
    c0 = wmma_bf16(a, b0, c0);
    c1 = wmma_bf16(a, b1, c1);
  }
  int n = lane & 15, hh = lane >> 4;
#pragma unroll
  for (int v = 0; v < 8; ++v) {
    ldsC[v + 8 * hh][n0 + n]      = c0[v];
    ldsC[v + 8 * hh][n0 + 16 + n] = c1[v];
  }
  if (t < 16) rowsq[t] = 0.f;
  __syncthreads();
  {
    int r = t >> 4, j = t & 15;
    float s = 0.f;
#pragma unroll
    for (int e = 0; e < 16; ++e) { float x = ldsC[r][j * 16 + e]; s += x * x; }
    atomicAdd(&rowsq[r], s);
  }
  __syncthreads();
  {
    int r = t >> 4, j = t & 15;
    float sc = rsqrtf(rowsq[r] * (1.f / DCQ_) + EPS_);
#pragma unroll
    for (int e = 0; e < 16; ++e) {
      int col = j * 16 + e;
      qc[(size_t)(m0 + r) * DCQ_ + col] = f2bf(nw[col] * ldsC[r][col] * sc);
    }
  }
}

// ---------------- q up-proj + RoPE ----------------
__global__ __launch_bounds__(192) void k_qup(const u16* __restrict__ qc,
                                             const u16* __restrict__ Wt,
                                             const int* __restrict__ qpos,
                                             u16* __restrict__ query) {
  __shared__ float lds[16][200];
  int head = blockIdx.x & 15, rb = blockIdx.x >> 4;
  int t = threadIdx.x, wave = t >> 5, lane = t & 31;
  int m0 = rb * 16;
  int n0 = head * QHD_ + wave * 32;
  f32x8 c0 = {}, c1 = {};
  for (int k0 = 0; k0 < DCQ_; k0 += 32) {
    Frag a, b0, b1;
    load_a(a, qc + (size_t)m0 * DCQ_ + k0, DCQ_, lane);
    load_b(b0, Wt + (size_t)n0 * DCQ_ + k0, DCQ_, lane);
    load_b(b1, Wt + (size_t)(n0 + 16) * DCQ_ + k0, DCQ_, lane);
    c0 = wmma_bf16(a, b0, c0);
    c1 = wmma_bf16(a, b1, c1);
  }
  int n = lane & 15, hh = lane >> 4, nl = wave * 32;
#pragma unroll
  for (int v = 0; v < 8; ++v) {
    lds[v + 8 * hh][nl + n]      = c0[v];
    lds[v + 8 * hh][nl + 16 + n] = c1[v];
  }
  __syncthreads();
  int d = t; // 0..191
  for (int r = 0; r < 16; ++r) {
    int row = m0 + r;
    int b = row >> 11, sq = row & (SQ_ - 1);
    float val;
    if (d < DPRE_) {
      val = lds[r][d];
    } else {
      int i = d - DPRE_;                    // 0..63
      int j = i & 31;
      float invf = __expf(-(float)j * (1.f / 32.f) * LN10K_);
      float ang = (float)qpos[row] * invf;
      float sn, cs; __sincosf(ang, &sn, &cs);
      int idx = (i < 32) ? 2 * i : 2 * (i - 32) + 1;
      float qr = lds[r][DPRE_ + idx];
      float rot = (i < 32) ? -lds[r][DPRE_ + 2 * i + 1] : lds[r][DPRE_ + 2 * (i - 32)];
      val = qr * cs + rot * sn;
    }
    query[(((size_t)(b * H_ + head)) * SQ_ + sq) * QHD_ + d] = f2bf(val);
  }
}

// ---------------- kv down-proj + RMS norm + RoPE(k_rope) ----------------
__global__ __launch_bounds__(160) void k_kvdown(const u16* __restrict__ A,
                                                const u16* __restrict__ Wt,
                                                const float* __restrict__ kvnw,
                                                const int* __restrict__ kvpos,
                                                u16* __restrict__ ckv,
                                                u16* __restrict__ krope) {
  __shared__ float lds[16][84];
  __shared__ float rowscale[16];
  int t = threadIdx.x, wave = t >> 5, lane = t & 31;
  int m0 = blockIdx.x * 16;
  int n0 = wave * 16;
  f32x8 c0 = {};
  for (int k0 = 0; k0 < D_; k0 += 32) {
    Frag a, b0;
    load_a(a, A + (size_t)m0 * D_ + k0, D_, lane);
    load_b(b0, Wt + (size_t)n0 * D_ + k0, D_, lane);
    c0 = wmma_bf16(a, b0, c0);
  }
  int n = lane & 15, hh = lane >> 4;
#pragma unroll
  for (int v = 0; v < 8; ++v) lds[v + 8 * hh][n0 + n] = c0[v];
  __syncthreads();
  if (t < 16) {
    float s = 0.f;
#pragma unroll
    for (int j = 0; j < DCKV_; ++j) { float x = lds[t][j]; s += x * x; }
    rowscale[t] = rsqrtf(s * (1.f / DCKV_) + EPS_);
  }
  __syncthreads();
  for (int idx = t; idx < 16 * DCKV_; idx += 160) {
    int r = idx >> 4, j = idx & 15;
    ckv[(size_t)(m0 + r) * DCKV_ + j] = f2bf(kvnw[j] * lds[r][j] * rowscale[r]);
  }
  for (int idx = t; idx < 16 * DROPE_; idx += 160) {
    int r = idx >> 6, i = idx & 63;
    int row = m0 + r;
    int j = i & 31;
    float invf = __expf(-(float)j * (1.f / 32.f) * LN10K_);
    float ang = (float)kvpos[row] * invf;
    float sn, cs; __sincosf(ang, &sn, &cs);
    int idn = (i < 32) ? 2 * i : 2 * (i - 32) + 1;
    float kr = lds[r][DCKV_ + idn];
    float rot = (i < 32) ? -lds[r][DCKV_ + 2 * i + 1] : lds[r][DCKV_ + 2 * (i - 32)];
    krope[(size_t)row * DROPE_ + i] = f2bf(kr * cs + rot * sn);
  }
}

// ---------------- kv up-proj ----------------
__global__ __launch_bounds__(256) void k_kvup(const u16* __restrict__ ckv,
                                              const u16* __restrict__ Wt,
                                              const u16* __restrict__ krope,
                                              u16* __restrict__ key,
                                              u16* __restrict__ valT) {
  int head = blockIdx.x & 15, rb = blockIdx.x >> 4;
  int t = threadIdx.x, wave = t >> 5, lane = t & 31;
  int m0 = rb * 16;
  int nl0 = wave * 32;
  Frag a, b0, b1;
  load_a_k16(a, ckv + (size_t)m0 * DCKV_, DCKV_, lane);
  load_b_k16(b0, Wt + (size_t)(head * MEGD_ + nl0) * DCKV_, DCKV_, lane);
  load_b_k16(b1, Wt + (size_t)(head * MEGD_ + nl0 + 16) * DCKV_, DCKV_, lane);
  f32x8 c0 = {}, c1 = {};
  c0 = wmma_bf16(a, b0, c0);
  c1 = wmma_bf16(a, b1, c1);
  int n = lane & 15, hh = lane >> 4;
#pragma unroll
  for (int v = 0; v < 8; ++v) {
    int m = m0 + v + 8 * hh;
    int b = m >> 11, skv = m & (SKV_ - 1);
    size_t bh = (size_t)(b * H_ + head);
    int nloc0 = nl0 + n, nloc1 = nl0 + 16 + n;
    if (nloc0 < DPRE_) key[(bh * SKV_ + skv) * QHD_ + nloc0] = f2bf(c0[v]);
    else               valT[(bh * DCV_ + (nloc0 - DPRE_)) * SKV_ + skv] = f2bf(c0[v]);
    if (nloc1 < DPRE_) key[(bh * SKV_ + skv) * QHD_ + nloc1] = f2bf(c1[v]);
    else               valT[(bh * DCV_ + (nloc1 - DPRE_)) * SKV_ + skv] = f2bf(c1[v]);
  }
  for (int idx = t; idx < 16 * DROPE_; idx += 256) {
    int r = idx >> 6, i = idx & 63;
    int m = m0 + r;
    int b = m >> 11, skv = m & (SKV_ - 1);
    key[(((size_t)(b * H_ + head)) * SKV_ + skv) * QHD_ + DPRE_ + i] =
        krope[(size_t)m * DROPE_ + i];
  }
}

// ---------------- flash attention ----------------
// grid = (B*H) * (SQ/128) = 512 ; block = 256 (8 waves, 16 q rows each).
// K/V tiles are staged ONCE per workgroup into LDS via async global->LDS
// (8x reduction in L2 traffic vs per-wave fragment loads), then all waves
// build WMMA B-fragments from LDS (two 16B runs per lane -> ds_load_b128 pairs).
__global__ __launch_bounds__(256) void k_attn(const u16* __restrict__ query,
                                              const u16* __restrict__ key,
                                              const u16* __restrict__ valT,
                                              u16* __restrict__ attn) {
  __shared__ __attribute__((aligned(16))) u16 Kt[32][200];  // 32 kv x 192 (+pad)
  __shared__ __attribute__((aligned(16))) u16 Vt[128][40];  // 128 dv x 32 (+pad)
  __shared__ __attribute__((aligned(16))) u16 Pl[8][16][32];
  int qblk = blockIdx.x & 15, bh = blockIdx.x >> 4;
  int b = bh >> 4, h = bh & 15;
  int t = threadIdx.x, wave = t >> 5, lane = t & 31;
  int q0 = qblk * 128 + wave * 16;
  int n = lane & 15, hh = lane >> 4;

  const u16* Qbase = query + ((size_t)bh * SQ_ + q0) * QHD_;
  Frag Qf[6];
#pragma unroll
  for (int d = 0; d < 6; ++d) load_a(Qf[d], Qbase + d * 32, QHD_, lane);

  float mstate[8], lstate[8];
#pragma unroll
  for (int v = 0; v < 8; ++v) { mstate[v] = -1e30f; lstate[v] = 0.f; }
  f32x8 acc[8];
#pragma unroll
  for (int i = 0; i < 8; ++i) acc[i] = (f32x8){};

  const u16* Kall = key + (size_t)bh * SKV_ * QHD_;
  const u16* Vall = valT + (size_t)bh * DCV_ * SKV_;
  u32 kt0 = (u32)(uintptr_t)(void*)&Kt[0][0];
  u32 vt0 = (u32)(uintptr_t)(void*)&Vt[0][0];

  for (int kv0 = 0; kv0 < SKV_; kv0 += 32) {
    const u16* Kbase = Kall + (size_t)kv0 * QHD_;
    __builtin_prefetch(Kbase + 32 * QHD_, 0, 1);
    __builtin_prefetch(Vall + kv0 + 32, 0, 1);

    // ---- cooperative stage: K tile 32x192 (768 x 16B), V tile 128x32 (512 x 16B)
#pragma unroll
    for (int j = 0; j < 3; ++j) {
      int idx = t + j * 256;                 // 0..767
      int row = idx / 24, c = idx - row * 24;
      const u16* src = Kbase + row * QHD_ + c * 8;
#if USE_ASYNC_COPY
      async_cp16(kt0 + (u32)(row * 200 + c * 8) * 2, src);
#else
      *(uint4*)&Kt[row][c * 8] = *(const uint4*)src;
#endif
    }
#pragma unroll
    for (int j = 0; j < 2; ++j) {
      int idx = t + j * 256;                 // 0..511
      int row = idx >> 2, c = idx & 3;
      const u16* src = Vall + (size_t)row * SKV_ + kv0 + c * 8;
#if USE_ASYNC_COPY
      async_cp16(vt0 + (u32)(row * 40 + c * 8) * 2, src);
#else
      *(uint4*)&Vt[row][c * 8] = *(const uint4*)src;
#endif
    }
#if USE_ASYNC_COPY
    asm volatile("s_wait_asynccnt 0x0" ::: "memory");
#endif
    __syncthreads();

    // ---- S = Q @ K^T (B-fragments straight from LDS rows)
    f32x8 s0 = {}, s1 = {};
#pragma unroll
    for (int d = 0; d < 6; ++d) {
      Frag b0, b1;
      load_b(b0, &Kt[0][0] + d * 32, 200, lane);
      load_b(b1, &Kt[16][0] + d * 32, 200, lane);
      s0 = wmma_bf16(Qf[d], b0, s0);
      s1 = wmma_bf16(Qf[d], b1, s1);
    }

    // ---- online softmax
    float corr[8];
#pragma unroll
    for (int v = 0; v < 8; ++v) {
      float a0 = s0[v] * ASCALE_, a1 = s1[v] * ASCALE_;
      float tm = fmaxf(a0, a1);
#pragma unroll
      for (int off = 1; off < 16; off <<= 1) tm = fmaxf(tm, __shfl_xor(tm, off, 16));
      float mn = fmaxf(mstate[v], tm);
      float p0 = __expf(a0 - mn), p1 = __expf(a1 - mn);
      float rs = p0 + p1;
#pragma unroll
      for (int off = 1; off < 16; off <<= 1) rs += __shfl_xor(rs, off, 16);
      float cr = __expf(mstate[v] - mn);
      lstate[v] = lstate[v] * cr + rs;
      mstate[v] = mn;
      corr[v] = cr;
      Pl[wave][v + 8 * hh][n]      = f2bf(p0);
      Pl[wave][v + 8 * hh][16 + n] = f2bf(p1);
    }
#pragma unroll
    for (int i = 0; i < 8; ++i)
#pragma unroll
      for (int v = 0; v < 8; ++v) acc[i][v] *= corr[v];

    asm volatile("s_wait_dscnt 0x0" ::: "memory"); // wave-private LDS store->load
    Frag pf;
    {
      int m = lane & 15, hx = lane >> 4;
#pragma unroll
      for (int v = 0; v < 4; ++v) pf.u[v]     = *(const u32*)&Pl[wave][m][hx * 8 + 2 * v];
#pragma unroll
      for (int v = 0; v < 4; ++v) pf.u[4 + v] = *(const u32*)&Pl[wave][m][16 + hx * 8 + 2 * v];
    }

    // ---- O += P @ V (V B-fragments from LDS)
#pragma unroll
    for (int i = 0; i < 8; ++i) {
      Frag bv;
      load_b(bv, &Vt[i * 16][0], 40, lane);
      acc[i] = wmma_bf16(pf, bv, acc[i]);
    }
    __syncthreads(); // before next tile overwrites Kt/Vt (and Pl)
  }

#pragma unroll
  for (int i = 0; i < 8; ++i) {
#pragma unroll
    for (int v = 0; v < 8; ++v) {
      int m = q0 + v + 8 * hh;
      float val = acc[i][v] / lstate[v];
      attn[((size_t)b * SQ_ + m) * OUTK_ + h * DCV_ + i * 16 + n] = f2bf(val);
    }
  }
}

// ---------------- output projection ----------------
__global__ __launch_bounds__(256) void k_outproj(const u16* __restrict__ attn,
                                                 const u16* __restrict__ Wt,
                                                 float* __restrict__ out) {
  int blkN = blockIdx.x & 3, blkM = blockIdx.x >> 2;
  int t = threadIdx.x, wave = t >> 5, lane = t & 31;
  int mw = wave >> 2, nw = wave & 3;
  int m0 = blkM * 32 + mw * 16;
  int n0 = blkN * 256 + nw * 64;
  f32x8 c[4];
#pragma unroll
  for (int j = 0; j < 4; ++j) c[j] = (f32x8){};
  for (int k0 = 0; k0 < OUTK_; k0 += 32) {
    Frag a;
    load_a(a, attn + (size_t)m0 * OUTK_ + k0, OUTK_, lane);
#pragma unroll
    for (int j = 0; j < 4; ++j) {
      Frag bb;
      load_b(bb, Wt + (size_t)(n0 + j * 16) * OUTK_ + k0, OUTK_, lane);
      c[j] = wmma_bf16(a, bb, c[j]);
    }
  }
  int n = lane & 15, hh = lane >> 4;
#pragma unroll
  for (int j = 0; j < 4; ++j)
#pragma unroll
    for (int v = 0; v < 8; ++v)
      out[(size_t)(m0 + v + 8 * hh) * D_ + n0 + j * 16 + n] = c[j][v];
}

// ---------------- host launcher ----------------
extern "C" void kernel_launch(void* const* d_in, const int* in_sizes, int n_in,
                              void* d_out, int out_size, void* d_ws, size_t ws_size,
                              hipStream_t stream) {
  const float* q_inputs = (const float*)d_in[0];
  const float* kv_input = (const float*)d_in[1];
  const int*   q_pos    = (const int*)d_in[2];
  const int*   kv_pos   = (const int*)d_in[3];
  const float* Wq_down  = (const float*)d_in[4];
  const float* q_norm_w = (const float*)d_in[5];
  const float* Wq_up    = (const float*)d_in[6];
  const float* Wkv_down = (const float*)d_in[7];
  const float* kv_norm_w= (const float*)d_in[8];
  const float* Wkv_up   = (const float*)d_in[9];
  const float* Wout     = (const float*)d_in[10];

  char* ws = (char*)d_ws;
  size_t off = 0;
  auto alloc = [&](size_t bytes) -> char* {
    char* p = ws + off;
    off = (off + bytes + 255) & ~(size_t)255;
    return p;
  };
  u16* qin_bf  = (u16*)alloc((size_t)BSQ_ * D_ * 2);
  u16* kvin_bf = (u16*)alloc((size_t)BSKV_ * D_ * 2);
  u16* WqdT    = (u16*)alloc((size_t)DCQ_ * D_ * 2);
  u16* WqupT   = (u16*)alloc((size_t)QUPN_ * DCQ_ * 2);
  u16* WkvdT   = (u16*)alloc((size_t)KVDN_ * D_ * 2);
  u16* WkvupT  = (u16*)alloc((size_t)KVUPN_ * DCKV_ * 2);
  u16* WoutT   = (u16*)alloc((size_t)D_ * OUTK_ * 2);
  u16* qc      = (u16*)alloc((size_t)BSQ_ * DCQ_ * 2);
  u16* query   = (u16*)alloc((size_t)B_ * H_ * SQ_ * QHD_ * 2);
  u16* ckv     = (u16*)alloc((size_t)BSKV_ * DCKV_ * 2);
  u16* krope   = (u16*)alloc((size_t)BSKV_ * DROPE_ * 2);
  u16* keyb    = (u16*)alloc((size_t)B_ * H_ * SKV_ * QHD_ * 2);
  u16* valT    = (u16*)alloc((size_t)B_ * H_ * DCV_ * SKV_ * 2);
  u16* attn    = (u16*)alloc((size_t)BSQ_ * OUTK_ * 2);
  (void)in_sizes; (void)n_in; (void)out_size; (void)ws_size;

  auto grid1 = [](size_t n) { return (unsigned)((n + 255) / 256); };

  k_cvt_bf16<<<grid1((size_t)BSQ_ * D_), 256, 0, stream>>>(q_inputs, qin_bf, BSQ_ * D_);
  k_cvt_bf16<<<grid1((size_t)BSKV_ * D_), 256, 0, stream>>>(kv_input, kvin_bf, BSKV_ * D_);
  k_cvt_tr<<<grid1((size_t)D_ * DCQ_), 256, 0, stream>>>(Wq_down, WqdT, D_, DCQ_);
  k_cvt_tr<<<grid1((size_t)DCQ_ * QUPN_), 256, 0, stream>>>(Wq_up, WqupT, DCQ_, QUPN_);
  k_cvt_tr<<<grid1((size_t)D_ * KVDN_), 256, 0, stream>>>(Wkv_down, WkvdT, D_, KVDN_);
  k_cvt_tr<<<grid1((size_t)DCKV_ * KVUPN_), 256, 0, stream>>>(Wkv_up, WkvupT, DCKV_, KVUPN_);
  k_cvt_tr<<<grid1((size_t)OUTK_ * D_), 256, 0, stream>>>(Wout, WoutT, OUTK_, D_);

  k_qdown<<<BSQ_ / 16, 256, 0, stream>>>(qin_bf, WqdT, q_norm_w, qc);
  k_qup<<<(BSQ_ / 16) * H_, 192, 0, stream>>>(qc, WqupT, q_pos, query);
  k_kvdown<<<BSKV_ / 16, 160, 0, stream>>>(kvin_bf, WkvdT, kv_norm_w, kv_pos, ckv, krope);
  k_kvup<<<(BSKV_ / 16) * H_, 256, 0, stream>>>(ckv, WkvupT, krope, keyb, valT);
  k_attn<<<(B_ * H_) * (SQ_ / 128), 256, 0, stream>>>(query, keyb, valT, attn);
  k_outproj<<<(BSQ_ / 32) * (D_ / 256), 256, 0, stream>>>(attn, WoutT, (float*)d_out);
}
